// DecoderLayer_15221364097644
// MI455X (gfx1250) — matrix operand order repaired and checked by reading, not compiled, not run on hardware
//
#include <hip/hip_runtime.h>
#include <hip/hip_bf16.h>

// ---------------------------------------------------------------------------
// Types for CDNA5 WMMA (wave32): 16x16x32 f16 -> f32
// ---------------------------------------------------------------------------
typedef __attribute__((ext_vector_type(16))) _Float16 v16h;
typedef __attribute__((ext_vector_type(8)))  _Float16 v8h;
typedef __attribute__((ext_vector_type(4)))  _Float16 v4h;
typedef __attribute__((ext_vector_type(8)))  float    v8f;

static __device__ inline v16h mk16(v8h lo, v8h hi) {
    v16h r;
#pragma unroll
    for (int i = 0; i < 8; ++i) { r[i] = lo[i]; r[i + 8] = hi[i]; }
    return r;
}

static __device__ inline v8f zero8f() {
    v8f z;
#pragma unroll
    for (int i = 0; i < 8; ++i) z[i] = 0.0f;
    return z;
}

// ---------------------------------------------------------------------------
// 16-lane xor-butterfly reductions via v_permlane16_b32 (VALU-only).
// ---------------------------------------------------------------------------
#if __has_builtin(__builtin_amdgcn_permlane16)
static __device__ inline float plx16(float v, unsigned sel0, unsigned sel1) {
    unsigned u = __builtin_bit_cast(unsigned, v);
    unsigned r = __builtin_amdgcn_permlane16(u, u, sel0, sel1, false, false);
    return __builtin_bit_cast(float, r);
}
static __device__ inline float rmax16(float v) {
    v = fmaxf(v, plx16(v, 0x67452301u, 0xEFCDAB89u));   // xor 1
    v = fmaxf(v, plx16(v, 0x54761032u, 0xDCFE98BAu));   // xor 2
    v = fmaxf(v, plx16(v, 0x32107654u, 0xBA98FEDCu));   // xor 4
    v = fmaxf(v, plx16(v, 0xFEDCBA98u, 0x76543210u));   // xor 8
    return v;
}
static __device__ inline float rsum16(float v) {
    v += plx16(v, 0x67452301u, 0xEFCDAB89u);
    v += plx16(v, 0x54761032u, 0xDCFE98BAu);
    v += plx16(v, 0x32107654u, 0xBA98FEDCu);
    v += plx16(v, 0xFEDCBA98u, 0x76543210u);
    return v;
}
#else
static __device__ inline float rmax16(float v) {
#pragma unroll
    for (int m = 1; m < 16; m <<= 1) v = fmaxf(v, __shfl_xor(v, m, 32));
    return v;
}
static __device__ inline float rsum16(float v) {
#pragma unroll
    for (int m = 1; m < 16; m <<= 1) v += __shfl_xor(v, m, 32);
    return v;
}
#endif

// ---------------------------------------------------------------------------
// Tiled WMMA GEMM: C[M,N] = A[M,K] @ B[K,N] + bias (opt relu)
// A: f32 or f16 (A_F16). B: f32 weights. Out: f32 or f16 (OUT_F16).
// OUT_TRANS: f16 output stored transposed per batch: [b][n][s] (S=2048).
// Block 256 threads = 8 waves (wave32). BM=BN=128, BK=64.
// Wave grid 2x4; each wave computes 64x32 via 4x2 WMMA accumulators.
// ---------------------------------------------------------------------------
template<bool A_F16, bool OUT_F16, bool RELU, bool OUT_TRANS>
__global__ __launch_bounds__(256) void gemm_wmma(
    const void* __restrict__ Ap, const float* __restrict__ Bp,
    const float* __restrict__ bias, void* __restrict__ Out,
    int M, int N, int K)
{
    __shared__ __align__(16) _Float16 As[128][72];   // [m][k], row pad 144B
    __shared__ __align__(16) _Float16 Bs[128][72];   // [n][k] (B transposed)

    const int tid  = threadIdx.x;
    const int lane = tid & 31;
    const int wave = tid >> 5;
    const int wr   = wave >> 2;       // 0..1
    const int wc   = wave & 3;        // 0..3
    const int g    = lane >> 4;       // half-wave group
    const int l16  = lane & 15;
    const int bm   = blockIdx.y * 128;
    const int bn   = blockIdx.x * 128;

    v8f acc[4][2];
#pragma unroll
    for (int i = 0; i < 4; ++i)
#pragma unroll
        for (int j = 0; j < 2; ++j) acc[i][j] = zero8f();

    const int nkb = K >> 6;
    for (int kb = 0; kb < nkb; ++kb) {
        const int k0 = kb << 6;

        // ---- stage A tile (128x64) as f16 ----
        if constexpr (A_F16) {
            const _Float16* A = (const _Float16*)Ap;
#pragma unroll
            for (int c = tid; c < 1024; c += 256) {       // 8-half chunks
                int r = c >> 3, col = (c & 7) << 3;
                v8h v = *(const v8h*)(A + (size_t)(bm + r) * K + k0 + col);
                *(v8h*)&As[r][col] = v;
            }
        } else {
            const float* A = (const float*)Ap;
#pragma unroll
            for (int c = tid; c < 2048; c += 256) {       // float4 chunks
                int r = c >> 4, col = (c & 15) << 2;
                float4 v = *(const float4*)(A + (size_t)(bm + r) * K + k0 + col);
                v4h h;
                h[0] = (_Float16)v.x; h[1] = (_Float16)v.y;
                h[2] = (_Float16)v.z; h[3] = (_Float16)v.w;
                *(v4h*)&As[r][col] = h;
            }
        }
        // ---- stage B tile (64x128) transposed: Bs[n][k] = B[k0+k][bn+n] ----
#pragma unroll
        for (int c = tid; c < 2048; c += 256) {
            int k = c >> 5, n4 = (c & 31) << 2;
            float4 v = *(const float4*)(Bp + (size_t)(k0 + k) * N + bn + n4);
            Bs[n4 + 0][k] = (_Float16)v.x;
            Bs[n4 + 1][k] = (_Float16)v.y;
            Bs[n4 + 2][k] = (_Float16)v.z;
            Bs[n4 + 3][k] = (_Float16)v.w;
        }

        // ---- prefetch next tiles into cache while we compute ----
        if (kb + 1 < nkb) {
            const int kn = k0 + 64;
            if constexpr (A_F16) {
                const _Float16* A = (const _Float16*)Ap;
                __builtin_prefetch(A + (size_t)(bm + (tid >> 1)) * K + kn + ((tid & 1) << 5), 0, 3);
            } else {
                const float* A = (const float*)Ap;
                __builtin_prefetch(A + (size_t)(bm + (tid >> 1)) * K + kn + ((tid & 1) << 5), 0, 3);
            }
            __builtin_prefetch(Bp + (size_t)(kn + (tid >> 2)) * N + bn + ((tid & 3) << 5), 0, 3);
        }
        __syncthreads();

        // ---- two K-steps of 32 from LDS ----
#pragma unroll
        for (int kk = 0; kk < 64; kk += 32) {
            v16h a[4], b[2];
#pragma unroll
            for (int mt = 0; mt < 4; ++mt) {
                int m = wr * 64 + mt * 16 + l16;
                v8h lo = *(const v8h*)&As[m][kk + g * 8];
                v8h hi = *(const v8h*)&As[m][kk + 16 + g * 8];
                a[mt] = mk16(lo, hi);
            }
#pragma unroll
            for (int nt = 0; nt < 2; ++nt) {
                int n = wc * 32 + nt * 16 + l16;
                v8h lo = *(const v8h*)&Bs[n][kk + g * 8];
                v8h hi = *(const v8h*)&Bs[n][kk + 16 + g * 8];
                b[nt] = mk16(lo, hi);
            }
#pragma unroll
            for (int mt = 0; mt < 4; ++mt)
#pragma unroll
                for (int nt = 0; nt < 2; ++nt)
                    acc[mt][nt] = __builtin_amdgcn_wmma_f32_16x16x32_f16(
                        false, a[mt], false, b[nt], (short)0, acc[mt][nt], false, false);
        }
        __syncthreads();
    }

    // ---- epilogue ----
#pragma unroll
    for (int nt = 0; nt < 2; ++nt) {
        int n = bn + wc * 32 + nt * 16 + l16;
        float bv = bias ? bias[n] : 0.0f;
#pragma unroll
        for (int mt = 0; mt < 4; ++mt) {
            int mbase = bm + wr * 64 + mt * 16 + g * 8;
#pragma unroll
            for (int i = 0; i < 8; ++i) {
                float v = acc[mt][nt][i] + bv;
                if (RELU) v = fmaxf(v, 0.0f);
                if constexpr (OUT_TRANS) {
                    // out[b][n][s] f16 : V^T layout; S fixed = 2048
                    int m  = mbase + i;
                    int bb = m >> 11;
                    int s  = m & 2047;
                    size_t off = (((size_t)bb * N + n) << 11) + s;
                    ((_Float16*)Out)[off] = (_Float16)v;
                } else {
                    size_t off = (size_t)(mbase + i) * N + n;
                    if constexpr (OUT_F16) ((_Float16*)Out)[off] = (_Float16)v;
                    else                   ((float*)Out)[off]    = v;
                }
            }
        }
    }
}

// ---------------------------------------------------------------------------
// Flash attention with causal mask + ALiBi. dk = 64. One wave per 16-query
// tile; block = 4 waves = 64 query rows of one (b,h). Key tiles of 64.
// Q/K: f16 [B,S,D].  Vt: f16 [B, D(=H*dk), S]  (transposed by V projection).
// ---------------------------------------------------------------------------
__global__ __launch_bounds__(128) void attn_wmma(
    const _Float16* __restrict__ Qg, const _Float16* __restrict__ Kg,
    const _Float16* __restrict__ Vt, _Float16* __restrict__ Og,
    int S, int D, int H)
{
    __shared__ __align__(16) _Float16 P[4][16][72];   // per-wave P staging (16x64)

    const int tid  = threadIdx.x;
    const int lane = tid & 31;
    const int wave = tid >> 5;           // 0..3
    const int g    = lane >> 4;
    const int l16  = lane & 15;
    const int bh   = blockIdx.y;
    const int b    = bh / H;
    const int h    = bh % H;
    const int qt0  = blockIdx.x * 64 + wave * 16;
    const int dk   = 64;
    const float scale = 0.125f;                         // 1/sqrt(64)
    const float slope = exp2f(-0.5f * (float)(h + 1));  // (2^(8/H))^-(h+1)

    const size_t rowstride = (size_t)D;
    const _Float16* qbase  = Qg + ((size_t)b * S) * rowstride + (size_t)h * dk;
    const _Float16* kbase  = Kg + ((size_t)b * S) * rowstride + (size_t)h * dk;
    const _Float16* vtbase = Vt + (size_t)b * D * S + (size_t)(h * dk) * S;

    // Q fragments (A layout): rows qt0..qt0+15, d = 0..63 split into 2 frags
    v16h qa[2];
    {
        const _Float16* qp = qbase + (size_t)(qt0 + l16) * rowstride;
#pragma unroll
        for (int f = 0; f < 2; ++f) {
            v8h lo = *(const v8h*)(qp + f * 32 + g * 8);
            v8h hi = *(const v8h*)(qp + f * 32 + 16 + g * 8);
            qa[f] = mk16(lo, hi);
        }
    }

    // ALiBi bias is affine in k0: bias(i,u,k0) = bi[i] + slope*16*u + slope*k0
    float rm[8], rs[8], bi[8];
    v8f o[4];
#pragma unroll
    for (int i = 0; i < 8; ++i) {
        rm[i] = -__builtin_inff();
        rs[i] = 0.0f;
        bi[i] = slope * (float)(l16 - (qt0 + i + g * 8));
    }
    const float du16 = slope * 16.0f;
#pragma unroll
    for (int w = 0; w < 4; ++w) o[w] = zero8f();

    for (int k0 = 0; k0 < qt0 + 16; k0 += 64) {
        const float bk = slope * (float)k0;

        // ---- preload all K fragments (16 b128 loads in flight) ----
        v16h kf[4][2];
#pragma unroll
        for (int u = 0; u < 4; ++u) {
            const _Float16* kp = kbase + (size_t)(k0 + u * 16 + l16) * rowstride;
#pragma unroll
            for (int f = 0; f < 2; ++f) {
                v8h lo = *(const v8h*)(kp + f * 32 + g * 8);
                v8h hi = *(const v8h*)(kp + f * 32 + 16 + g * 8);
                kf[u][f] = mk16(lo, hi);
            }
        }

        // ---- S = Q @ K^T for a 16x64 key tile (4 sub-tiles of 16 keys) ----
        v8f s[4];
#pragma unroll
        for (int u = 0; u < 4; ++u) s[u] = zero8f();
#pragma unroll
        for (int u = 0; u < 4; ++u)
#pragma unroll
            for (int f = 0; f < 2; ++f)
                s[u] = __builtin_amdgcn_wmma_f32_16x16x32_f16(
                    false, qa[f], false, kf[u][f], (short)0, s[u], false, false);

        // ---- scale + ALiBi + causal mask; online softmax; stage P in LDS ----
#pragma unroll
        for (int i = 0; i < 8; ++i) {
            const int qrow = qt0 + i + g * 8;
            float vu[4];
            float best = -__builtin_inff();
#pragma unroll
            for (int u = 0; u < 4; ++u) {
                const int key = k0 + u * 16 + l16;
                float v = fmaf(s[u][i], scale, bi[i] + du16 * (float)u + bk);
                if (key > qrow) v = -1e30f;
                vu[u] = v;
                best = fmaxf(best, v);
            }
            best = rmax16(best);

            const float mnew  = fmaxf(rm[i], best);
            const float alpha = __expf(rm[i] - mnew);
            rm[i] = mnew;
            float lsum = 0.0f;
#pragma unroll
            for (int u = 0; u < 4; ++u) {
                float p = __expf(vu[u] - mnew);
                P[wave][i + g * 8][u * 16 + l16] = (_Float16)p;
                lsum += p;
            }
            lsum = rsum16(lsum);
            rs[i] = rs[i] * alpha + lsum;
#pragma unroll
            for (int w = 0; w < 4; ++w) o[w][i] *= alpha;
        }

        // ---- read P back as A fragments ----
        v16h pa[2];
#pragma unroll
        for (int f = 0; f < 2; ++f) {
            v8h lo = *(const v8h*)&P[wave][l16][f * 32 + g * 8];
            v8h hi = *(const v8h*)&P[wave][l16][f * 32 + 16 + g * 8];
            pa[f] = mk16(lo, hi);
        }

        // ---- preload all V fragments, then O += P @ V ----
        v16h vf[4][2];
#pragma unroll
        for (int w = 0; w < 4; ++w) {
            const _Float16* vp = vtbase + (size_t)(w * 16 + l16) * S + k0;
#pragma unroll
            for (int f = 0; f < 2; ++f) {
                v8h lo = *(const v8h*)(vp + f * 32 + g * 8);
                v8h hi = *(const v8h*)(vp + f * 32 + 16 + g * 8);
                vf[w][f] = mk16(lo, hi);
            }
        }
#pragma unroll
        for (int w = 0; w < 4; ++w)
#pragma unroll
            for (int f = 0; f < 2; ++f)
                o[w] = __builtin_amdgcn_wmma_f32_16x16x32_f16(
                    false, pa[f], false, vf[w][f], (short)0, o[w], false, false);
    }

    // ---- normalize and store (f16, [B,S,D] with head slice) ----
    float rinv[8];
#pragma unroll
    for (int i = 0; i < 8; ++i) rinv[i] = 1.0f / rs[i];

    _Float16* ob = Og + ((size_t)b * S) * rowstride + (size_t)h * dk;
#pragma unroll
    for (int w = 0; w < 4; ++w) {
        const int d = w * 16 + l16;
#pragma unroll
        for (int i = 0; i < 8; ++i) {
            const int row = qt0 + i + g * 8;
            ob[(size_t)row * rowstride + d] = (_Float16)(o[w][i] * rinv[i]);
        }
    }
}

// ---------------------------------------------------------------------------
// out = LayerNorm(x + r) * gamma + beta     (D = 1024, one block per row)
// ---------------------------------------------------------------------------
__global__ __launch_bounds__(256) void add_ln(
    const float* __restrict__ x, const float* __restrict__ r,
    const float* __restrict__ gamma, const float* __restrict__ beta,
    float* __restrict__ out)
{
    const int Dc = 1024;
    __shared__ float red[16];
    const int row = blockIdx.x;
    const int tid = threadIdx.x;
    const float* xp = x + (size_t)row * Dc;
    const float* rp = r + (size_t)row * Dc;
    float* op = out + (size_t)row * Dc;

    float vals[4];
    float sum = 0.0f, sq = 0.0f;
#pragma unroll
    for (int i = 0; i < 4; ++i) {
        int idx = tid + i * 256;
        float v = xp[idx] + rp[idx];
        vals[i] = v;
        sum += v; sq += v * v;
    }
    sum = rsum16(sum); sq = rsum16(sq);
    sum += __shfl_xor(sum, 16, 32);
    sq  += __shfl_xor(sq,  16, 32);
    const int wave = tid >> 5, lane = tid & 31;
    if (lane == 0) { red[wave] = sum; red[8 + wave] = sq; }
    __syncthreads();
    if (wave == 0) {
        float s = (lane < 8) ? red[lane] : 0.0f;
        float q = (lane < 8) ? red[8 + lane] : 0.0f;
#pragma unroll
        for (int m = 1; m < 8; m <<= 1) {
            s += __shfl_xor(s, m, 32);
            q += __shfl_xor(q, m, 32);
        }
        if (lane == 0) { red[0] = s; red[1] = q; }
    }
    __syncthreads();
    const float mu  = red[0] * (1.0f / Dc);
    const float var = red[1] * (1.0f / Dc) - mu * mu;
    const float inv = rsqrtf(var + 1e-5f);
#pragma unroll
    for (int i = 0; i < 4; ++i) {
        int idx = tid + i * 256;
        op[idx] = (vals[i] - mu) * inv * gamma[idx] + beta[idx];
    }
}

// ---------------------------------------------------------------------------
// Host launcher
// ---------------------------------------------------------------------------
extern "C" void kernel_launch(void* const* d_in, const int* in_sizes, int n_in,
                              void* d_out, int out_size, void* d_ws, size_t ws_size,
                              hipStream_t stream)
{
    (void)in_sizes; (void)n_in; (void)out_size; (void)ws_size;
    const int Bc = 2, Sc = 2048, Dc = 1024, Hc = 16, DFFc = 4096;
    const int M = Bc * Sc;                       // 4096 rows

    const float* x   = (const float*)d_in[0];
    // d_in[1] = causal mask (recomputed analytically) -- unused
    const float* Wq  = (const float*)d_in[2];  const float* bq  = (const float*)d_in[3];
    const float* Wk  = (const float*)d_in[4];  const float* bk  = (const float*)d_in[5];
    const float* Wv  = (const float*)d_in[6];  const float* bv  = (const float*)d_in[7];
    const float* Wo  = (const float*)d_in[8];  const float* bo  = (const float*)d_in[9];
    const float* W1  = (const float*)d_in[10]; const float* b1  = (const float*)d_in[11];
    const float* W2  = (const float*)d_in[12]; const float* b2  = (const float*)d_in[13];
    const float* g1  = (const float*)d_in[14]; const float* be1 = (const float*)d_in[15];
    const float* g2  = (const float*)d_in[16]; const float* be2 = (const float*)d_in[17];

    // workspace layout (96 MB total)
    char* ws = (char*)d_ws;
    _Float16* q16  = (_Float16*)(ws);                          //  8 MB
    _Float16* k16  = (_Float16*)(ws + (size_t)8  * (1u << 20));//  8 MB
    _Float16* vt16 = (_Float16*)(ws + (size_t)16 * (1u << 20));//  8 MB (V^T)
    _Float16* a16  = (_Float16*)(ws + (size_t)24 * (1u << 20));//  8 MB
    float*    proj = (float*)   (ws + (size_t)32 * (1u << 20));// 16 MB (reused)
    float*    h1   = (float*)   (ws + (size_t)48 * (1u << 20));// 16 MB
    _Float16* ffh  = (_Float16*)(ws + (size_t)64 * (1u << 20));// 32 MB

    dim3 blk(256);
    dim3 gD(Dc / 128, M / 128);       // (8, 32)
    dim3 gF(DFFc / 128, M / 128);     // (32, 32)

    // QKV projections (f32 in, f16 out); V written transposed [B, D, S]
    gemm_wmma<false, true, false, false><<<gD, blk, 0, stream>>>(x, Wq, bq, q16,  M, Dc, Dc);
    gemm_wmma<false, true, false, false><<<gD, blk, 0, stream>>>(x, Wk, bk, k16,  M, Dc, Dc);
    gemm_wmma<false, true, false, true ><<<gD, blk, 0, stream>>>(x, Wv, bv, vt16, M, Dc, Dc);

    // Flash attention (causal + ALiBi), f16 out
    attn_wmma<<<dim3(Sc / 64, Bc * Hc), dim3(128), 0, stream>>>(q16, k16, vt16, a16, Sc, Dc, Hc);

    // Output projection (f16 in, f32 out), then residual + LN1
    gemm_wmma<true, false, false, false><<<gD, blk, 0, stream>>>(a16, Wo, bo, proj, M, Dc, Dc);
    add_ln<<<dim3(M), blk, 0, stream>>>(x, proj, g1, be1, h1);

    // FFN: relu(h1 @ W1 + b1) @ W2 + b2, then residual + LN2 -> d_out
    gemm_wmma<false, true, true,  false><<<gF, blk, 0, stream>>>(h1, W1, b1, ffh, M, DFFc, Dc);
    gemm_wmma<true, false, false, false><<<gD, blk, 0, stream>>>(ffh, W2, b2, proj, M, Dc, DFFc);
    add_ln<<<dim3(M), blk, 0, stream>>>(h1, proj, g2, be2, (float*)d_out);
}
